// QuantizedLayer_24163486007516
// MI455X (gfx1250) — matrix operand order repaired and checked by reading, not compiled, not run on hardware
//
#include <hip/hip_runtime.h>
#include <stdint.h>

typedef __attribute__((ext_vector_type(16))) __bf16 v16bf;
typedef __attribute__((ext_vector_type(8)))  float  v8f;
typedef unsigned int v4u __attribute__((ext_vector_type(4)));
typedef int          v8i __attribute__((ext_vector_type(8)));
typedef int          v4i __attribute__((ext_vector_type(4)));

#define OUT_F 4096
#define IN_F  4096
#define MROWS 8192      // B * S = 4 * 2048
#define BM 128
#define BN 128

// ---- fused-path tile config (fallback) ----
#define BK 32
#define LDS_STRIDE 40   // 32 bf16 + 8 pad; 80 B/row

// ---- TDM-path tile config ----
#define BKT 64          // bf16 K per tile
#define TSTRIDE 72      // 64 bf16 + 8 bf16 TDM pad => 144 B/row, 16B aligned
#define TILE_US (BM * TSTRIDE)          // ushorts per LDS tile buffer (9216)
#define TILE_BYTES (TILE_US * 2)        // 18432 B

#if __has_include(<hip/amd_detail/amd_gfx1250_TDM.h>)
#define CDNA5_TDM_6ARG 1
#else
#define CDNA5_TDM_6ARG 0
#endif

union Frag {
    v16bf v;
    uint4 q[2];
};

__device__ __forceinline__ unsigned short f32_to_bf16(float f) {
    union { float f; unsigned u; } c;
    c.f = f;
    unsigned u = c.u;
    u += 0x7FFFu + ((u >> 16) & 1u);   // round-to-nearest-even
    return (unsigned short)(u >> 16);
}

// ---------------------------------------------------------------------------
// TDM 2D tile load: global (row-major, elem=bf16) -> LDS with per-row padding.
// D# per CDNA5 ISA ch.8: group0 {count,lds_addr,global_addr,type=2},
// group1 {data_size=2B, pad 4 DW every 32 DW, dims/strides}.
// ---------------------------------------------------------------------------
__device__ __forceinline__ void tdm_load_tile_2d(
    unsigned lds_byte_off,               // LDS byte address of tile start
    const unsigned short* gptr,          // global tile start
    unsigned tensor_d0, unsigned tensor_d1,
    unsigned tile_d0, unsigned tile_d1,
    unsigned row_stride_elems)
{
    unsigned long long ga = (unsigned long long)(uintptr_t)gptr;
    v4u g0;
    g0[0] = 1u;                                          // count=1, user mode
    g0[1] = lds_byte_off;                                // lds_addr
    g0[2] = (unsigned)(ga & 0xFFFFFFFFu);                // global_addr[31:0]
    g0[3] = (unsigned)((ga >> 32) & 0x01FFFFFFu) | 0x80000000u;  // [56:32] | type=2

    v8i g1;
    g1[0] = (int)((1u << 16)        // data_size = 1 -> 2 bytes
                | (1u << 20)        // pad_enable
                | (4u << 22)        // pad_interval: 32 DWORDs (one 128B row)
                | (3u << 25));      // pad_amount: 4 DWORDs (16 B)
    g1[1] = (int)((tensor_d0 & 0xFFFFu) << 16);                      // dim0[15:0]
    g1[2] = (int)(((tensor_d0 >> 16) & 0xFFFFu) |
                  ((tensor_d1 & 0xFFFFu) << 16));                    // dim0[31:16]|dim1[15:0]
    g1[3] = (int)(((tensor_d1 >> 16) & 0xFFFFu) |
                  ((tile_d0 & 0xFFFFu) << 16));                      // dim1[31:16]|tile0
    g1[4] = (int)(tile_d1 & 0xFFFFu);                                // tile1, tile2=0
    g1[5] = (int)row_stride_elems;                                   // dim0_stride[31:0]
    g1[6] = 0;
    g1[7] = 0;

    v4i z = {0, 0, 0, 0};
#if CDNA5_TDM_6ARG
    v8i z8 = {0, 0, 0, 0, 0, 0, 0, 0};
    __builtin_amdgcn_tensor_load_to_lds(g0, g1, z, z, z8, 0);
#else
    __builtin_amdgcn_tensor_load_to_lds(g0, g1, z, z, 0);
#endif
}

// ---------------------------------------------------------------------------
// Kernel 1: dequantize weight indices -> bf16 weight matrix (one pass)
// ---------------------------------------------------------------------------
__global__ __launch_bounds__(256) void dequant_w_bf16(
    const int* __restrict__ widx, const float* __restrict__ centroid,
    unsigned short* __restrict__ wbf)
{
    __shared__ float cent[256];
    cent[threadIdx.x] = centroid[threadIdx.x];
    __syncthreads();

    size_t base = ((size_t)blockIdx.x * 256 + threadIdx.x) * 8;
    const int4* s = (const int4*)(widx + base);
    int4 a = s[0], b = s[1];
    unsigned short t[8];
    t[0] = f32_to_bf16(cent[a.x]); t[1] = f32_to_bf16(cent[a.y]);
    t[2] = f32_to_bf16(cent[a.z]); t[3] = f32_to_bf16(cent[a.w]);
    t[4] = f32_to_bf16(cent[b.x]); t[5] = f32_to_bf16(cent[b.y]);
    t[6] = f32_to_bf16(cent[b.z]); t[7] = f32_to_bf16(cent[b.w]);
    *(uint4*)(wbf + base) = *(const uint4*)t;
}

// ---------------------------------------------------------------------------
// Kernel 2: fp32 activations -> bf16 (one pass)
// ---------------------------------------------------------------------------
__global__ __launch_bounds__(256) void cvt_x_bf16(
    const float* __restrict__ x, unsigned short* __restrict__ xbf)
{
    size_t base = ((size_t)blockIdx.x * 256 + threadIdx.x) * 8;
    const float4* s = (const float4*)(x + base);
    float4 a = s[0], b = s[1];
    unsigned short t[8];
    t[0] = f32_to_bf16(a.x); t[1] = f32_to_bf16(a.y);
    t[2] = f32_to_bf16(a.z); t[3] = f32_to_bf16(a.w);
    t[4] = f32_to_bf16(b.x); t[5] = f32_to_bf16(b.y);
    t[6] = f32_to_bf16(b.z); t[7] = f32_to_bf16(b.w);
    *(uint4*)(xbf + base) = *(const uint4*)t;
}

// ---------------------------------------------------------------------------
// Kernel 3: bf16 GEMM, tiles staged by the Tensor Data Mover, double buffered.
// LDS layout (dynamic, so TDM lds_addr offsets are compile-time):
//   A0 @ 0, A1 @ TILE_BYTES, B0 @ 2*TILE_BYTES, B1 @ 3*TILE_BYTES
// ---------------------------------------------------------------------------
__global__ __launch_bounds__(256) void qgemm_tdm_bf16(
    const unsigned short* __restrict__ xbf,   // [MROWS, IN_F] bf16
    const unsigned short* __restrict__ wbf,   // [OUT_F, IN_F] bf16
    float* __restrict__ out)                  // [MROWS, OUT_F] fp32
{
    extern __shared__ __align__(16) unsigned short smem[];

    const int tid  = threadIdx.x;
    const int lane = tid & 31;
    const int wave = tid >> 5;
    const int wr   = wave & 3;   // fragment rows {2*wr, 2*wr+1}
    const int wc   = wave >> 2;  // fragment cols {4*wc .. 4*wc+3}

    const int m0 = blockIdx.x * BM;
    const int n0 = blockIdx.y * BN;

    const unsigned short* aBase = xbf + (size_t)m0 * IN_F;
    const unsigned short* bBase = wbf + (size_t)n0 * IN_F;

    const int NT = IN_F / BKT;   // 64 K-tiles

    if (wave == 0) {
        tdm_load_tile_2d(0u,             aBase, IN_F, MROWS, BKT, BM, IN_F);
        tdm_load_tile_2d(2u * TILE_BYTES, bBase, IN_F, OUT_F, BKT, BN, IN_F);
    }

    v8f acc[2][4] = {};

    const int lm   = lane & 15;
    const int aOff = (lane < 16) ? 0 : 8;    // A: K-octet select per half-wave
    const int bOff = (lane < 16) ? 0 : 8;    // B: K-octet select per half-wave

    for (int t = 0; t < NT; ++t) {
        const int buf = t & 1;
        if (wave == 0) {
            if (t + 1 < NT) {
                const unsigned short* an = aBase + (size_t)(t + 1) * BKT;
                const unsigned short* bn = bBase + (size_t)(t + 1) * BKT;
                const unsigned nb = (unsigned)(buf ^ 1) * TILE_BYTES;
                tdm_load_tile_2d(nb,                 an, IN_F, MROWS, BKT, BM, IN_F);
                tdm_load_tile_2d(2u * TILE_BYTES + nb, bn, IN_F, OUT_F, BKT, BN, IN_F);
                __builtin_amdgcn_s_wait_tensorcnt((short)2);  // tile t complete
            } else {
                __builtin_amdgcn_s_wait_tensorcnt((short)0);
            }
        }
        __syncthreads();

        const unsigned short* At = smem + buf * TILE_US;
        const unsigned short* Bt = smem + (2 + buf) * TILE_US;

#pragma unroll
        for (int ks = 0; ks < 2; ++ks) {   // two 16x16x32 K-steps per TDM tile
            Frag afr[2];
#pragma unroll
            for (int i = 0; i < 2; ++i) {
                const unsigned short* p =
                    &At[((wr * 2 + i) * 16 + lm) * TSTRIDE + ks * 32 + aOff];
                afr[i].q[0] = *(const uint4*)p;
                afr[i].q[1] = *(const uint4*)(p + 16);
            }
            Frag bfr[4];
#pragma unroll
            for (int j = 0; j < 4; ++j) {
                const unsigned short* p =
                    &Bt[((wc * 4 + j) * 16 + lm) * TSTRIDE + ks * 32 + bOff];
                bfr[j].q[0] = *(const uint4*)p;
                bfr[j].q[1] = *(const uint4*)(p + 16);
            }
#pragma unroll
            for (int i = 0; i < 2; ++i)
#pragma unroll
                for (int j = 0; j < 4; ++j)
                    acc[i][j] = __builtin_amdgcn_wmma_f32_16x16x32_bf16(
                        false, afr[i].v, false, bfr[j].v,
                        (short)0, acc[i][j], false, false);
        }
        __syncthreads();   // all waves done reading before buf is re-filled
    }

    const int rsel = (lane >> 4) * 8;
#pragma unroll
    for (int i = 0; i < 2; ++i) {
        const int mfrag = m0 + (wr * 2 + i) * 16 + rsel;
#pragma unroll
        for (int j = 0; j < 4; ++j) {
            const int n = n0 + (wc * 4 + j) * 16 + lm;
            float* o = out + (size_t)mfrag * OUT_F + n;
#pragma unroll
            for (int e = 0; e < 8; ++e)
                o[(size_t)e * OUT_F] = acc[i][j][e];
        }
    }
}

// ---------------------------------------------------------------------------
// Fallback: fused dequant + GEMM (no workspace requirement)
// ---------------------------------------------------------------------------
__global__ __launch_bounds__(256) void qlinear_wmma_fused(
    const float* __restrict__ input, const float* __restrict__ centroid,
    const int* __restrict__ widx, float* __restrict__ out)
{
    __shared__ __align__(16) unsigned short Atile[BM * LDS_STRIDE];
    __shared__ __align__(16) unsigned short Btile[BN * LDS_STRIDE];
    __shared__ float cent[256];

    const int tid  = threadIdx.x;
    const int lane = tid & 31;
    const int wave = tid >> 5;
    const int wr   = wave & 3;
    const int wc   = wave >> 2;

    const int m0 = blockIdx.x * BM;
    const int n0 = blockIdx.y * BN;

    cent[tid] = centroid[tid];

    const int row = tid >> 1;
    const int kh  = (tid & 1) * 16;
    const float* aSrcBase = input + (size_t)(m0 + row) * IN_F + kh;
    const int*   bSrcBase = widx  + (size_t)(n0 + row) * IN_F + kh;
    unsigned short* aDst = &Atile[row * LDS_STRIDE + kh];
    unsigned short* bDst = &Btile[row * LDS_STRIDE + kh];

    v8f acc[2][4] = {};
    __syncthreads();

    for (int kt = 0; kt < IN_F; kt += BK) {
        if (kt + BK < IN_F) {
            __builtin_prefetch(aSrcBase + kt + BK, 0, 1);
            __builtin_prefetch(bSrcBase + kt + BK, 0, 1);
        }
        {
            const float4* a4 = (const float4*)(aSrcBase + kt);
            unsigned short tmp[16];
#pragma unroll
            for (int v = 0; v < 4; ++v) {
                float4 f = a4[v];
                tmp[v * 4 + 0] = f32_to_bf16(f.x);
                tmp[v * 4 + 1] = f32_to_bf16(f.y);
                tmp[v * 4 + 2] = f32_to_bf16(f.z);
                tmp[v * 4 + 3] = f32_to_bf16(f.w);
            }
            ((uint4*)aDst)[0] = *(const uint4*)&tmp[0];
            ((uint4*)aDst)[1] = *(const uint4*)&tmp[8];
        }
        {
            const int4* b4 = (const int4*)(bSrcBase + kt);
            unsigned short tmp[16];
#pragma unroll
            for (int v = 0; v < 4; ++v) {
                int4 ix = b4[v];
                tmp[v * 4 + 0] = f32_to_bf16(cent[ix.x]);
                tmp[v * 4 + 1] = f32_to_bf16(cent[ix.y]);
                tmp[v * 4 + 2] = f32_to_bf16(cent[ix.z]);
                tmp[v * 4 + 3] = f32_to_bf16(cent[ix.w]);
            }
            ((uint4*)bDst)[0] = *(const uint4*)&tmp[0];
            ((uint4*)bDst)[1] = *(const uint4*)&tmp[8];
        }
        __syncthreads();

        const int lm   = lane & 15;
        const int aOff = (lane < 16) ? 0 : 8;
        const int bOff = (lane < 16) ? 0 : 8;

        Frag afr[2];
#pragma unroll
        for (int i = 0; i < 2; ++i) {
            const unsigned short* p =
                &Atile[((wr * 2 + i) * 16 + lm) * LDS_STRIDE + aOff];
            afr[i].q[0] = *(const uint4*)p;
            afr[i].q[1] = *(const uint4*)(p + 16);
        }
        Frag bfr[4];
#pragma unroll
        for (int j = 0; j < 4; ++j) {
            const unsigned short* p =
                &Btile[((wc * 4 + j) * 16 + lm) * LDS_STRIDE + bOff];
            bfr[j].q[0] = *(const uint4*)p;
            bfr[j].q[1] = *(const uint4*)(p + 16);
        }
#pragma unroll
        for (int i = 0; i < 2; ++i)
#pragma unroll
            for (int j = 0; j < 4; ++j)
                acc[i][j] = __builtin_amdgcn_wmma_f32_16x16x32_bf16(
                    false, afr[i].v, false, bfr[j].v,
                    (short)0, acc[i][j], false, false);
        __syncthreads();
    }

    const int lm   = lane & 15;
    const int rsel = (lane >> 4) * 8;
#pragma unroll
    for (int i = 0; i < 2; ++i) {
        const int mfrag = m0 + (wr * 2 + i) * 16 + rsel;
#pragma unroll
        for (int j = 0; j < 4; ++j) {
            const int n = n0 + (wc * 4 + j) * 16 + lm;
            float* o = out + (size_t)mfrag * OUT_F + n;
#pragma unroll
            for (int e = 0; e < 8; ++e)
                o[(size_t)e * OUT_F] = acc[i][j][e];
        }
    }
}

extern "C" void kernel_launch(void* const* d_in, const int* in_sizes, int n_in,
                              void* d_out, int out_size, void* d_ws, size_t ws_size,
                              hipStream_t stream) {
    const float* input    = (const float*)d_in[0];   // [4,2048,4096] fp32
    const float* centroid = (const float*)d_in[1];   // [256,1] fp32
    const int*   widx     = (const int*)d_in[2];     // [4096,4096] int32
    float*       out      = (float*)d_out;           // [4,2048,4096] fp32

    const size_t wbytes = (size_t)OUT_F * IN_F * 2;   // 32 MiB bf16 weights
    const size_t xbytes = (size_t)MROWS * IN_F * 2;   // 64 MiB bf16 activations
    dim3 grid(MROWS / BM, OUT_F / BN);                // 64 x 32 workgroups

    if (ws_size >= wbytes + xbytes) {
        unsigned short* wbf = (unsigned short*)d_ws;
        unsigned short* xbf = (unsigned short*)((char*)d_ws + wbytes);
        dequant_w_bf16<<<(OUT_F * IN_F) / (256 * 8), 256, 0, stream>>>(widx, centroid, wbf);
        cvt_x_bf16<<<(MROWS * IN_F) / (256 * 8), 256, 0, stream>>>(input, xbf);
        qgemm_tdm_bf16<<<grid, 256, 4 * TILE_BYTES, stream>>>(xbf, wbf, out);
    } else {
        qlinear_wmma_fused<<<grid, 256, 0, stream>>>(input, centroid, widx, out);
    }
}
